// DPF_EROT_53085795778861
// MI455X (gfx1250) — matrix operand order repaired and checked by reading, not compiled, not run on hardware
//
#include <hip/hip_runtime.h>
#include <stdint.h>

// Problem constants (reference: n=8192, d=64, eps=100, 50 Sinkhorn iters)
#define NPART 8192
#define DIMS  64
#define EPSR  100.0f
#define LOG2E 1.4426950408889634f
#define SHIFT2 29.0f   // exponent shift: K is used only up to per-row / global scale

typedef __attribute__((ext_vector_type(16))) __bf16 bf16x16;
typedef __attribute__((ext_vector_type(8)))  float  f32x8;

__device__ __forceinline__ unsigned short f2bf(float f) {
  unsigned u = __float_as_uint(f);
  return (unsigned short)((u + 0x7FFFu + ((u >> 16) & 1u)) >> 16); // RNE
}
__device__ __forceinline__ float bflo(unsigned u) { return __uint_as_float(u << 16); }
__device__ __forceinline__ float bfhi(unsigned u) { return __uint_as_float(u & 0xFFFF0000u); }

// CDNA5 async global->LDS copy (ASYNCcnt-tracked, no VGPR data path).
// GVS mode: mem = SGPR64 base + VGPR32 byte offset; LDS dest = VGPR32 offset
// (low 32 bits of a generic pointer to __shared__ are the LDS byte address).
__device__ __forceinline__ void async_copy_b128(void* lds_dst, const void* gbase,
                                                unsigned byte_off) {
  unsigned lds_off = (unsigned)(uintptr_t)lds_dst;
  asm volatile("global_load_async_to_lds_b128 %0, %1, %2"
               :: "v"(lds_off), "v"(byte_off), "s"(gbase)
               : "memory");
}
__device__ __forceinline__ void async_wait0() {
  asm volatile("s_wait_asynccnt 0x0" ::: "memory");
}

// ---------------------------------------------------------------------------
// init: per-row sq-norm, c_i = log2e*(-sq/eps + log w_i), f32->bf16 particles,
//       v0 = 1.0 (with S[0]=n this is v0=1/n), one wave (32 lanes) per row.
// ---------------------------------------------------------------------------
__global__ void __launch_bounds__(256) dpf_init(const float* __restrict__ P,
                                                const float* __restrict__ w,
                                                unsigned short* __restrict__ Pbf,
                                                float* __restrict__ c,
                                                float* __restrict__ v,
                                                float* __restrict__ S) {
  int t = threadIdx.x, wave = t >> 5, lane = t & 31;
  int row = blockIdx.x * 8 + wave;
  float2 p = *(const float2*)&P[row * DIMS + lane * 2];
  float sq = p.x * p.x + p.y * p.y;
  #pragma unroll
  for (int off = 16; off; off >>= 1) sq += __shfl_down(sq, off, 32);
  unsigned packed = ((unsigned)f2bf(p.y) << 16) | (unsigned)f2bf(p.x);
  *(unsigned*)&Pbf[row * DIMS + lane * 2] = packed;
  if (lane == 0) {
    c[row] = LOG2E * (-sq / EPSR + __logf(w[row]));
    v[row] = 1.0f;
  }
  if (blockIdx.x == 0 && t == 0) S[0] = (float)NPART;
}

// ---------------------------------------------------------------------------
// build_K: K_ij = exp2( (2/eps)*log2e * (P_i . P_j) + c_i + c_j + SHIFT2 )
// 128x128 tile per 256-thread block; Gram via v_wmma_f32_16x16x32_bf16.
// Per-wave LDS staging tile turns the C fragment into coalesced b128 stores.
// ---------------------------------------------------------------------------
__global__ void __launch_bounds__(256) dpf_build_k(const unsigned short* __restrict__ Pbf,
                                                   const float* __restrict__ c,
                                                   unsigned short* __restrict__ K) {
  __shared__ __align__(16) unsigned short As[128 * DIMS];
  __shared__ __align__(16) unsigned short Bs[128 * DIMS];
  __shared__ __align__(16) unsigned short stage[8][256]; // per-wave 16x16 bf16 tile
  __shared__ float cA[128], cB[128];

  int i0 = blockIdx.y * 128, j0 = blockIdx.x * 128;
  int t = threadIdx.x;
  #pragma unroll
  for (int it = 0; it < 4; ++it) {
    int idx = (it * 256 + t) * 8; // half index, 16B aligned
    async_copy_b128(&As[idx], Pbf, (unsigned)(i0 * DIMS + idx) * 2u);
    async_copy_b128(&Bs[idx], Pbf, (unsigned)(j0 * DIMS + idx) * 2u);
  }
  if (t < 128) { cA[t] = c[i0 + t]; cB[t] = c[j0 + t]; }
  async_wait0();
  __syncthreads();

  int wave = t >> 5, lane = t & 31;
  int lm = lane & 15, lh = lane >> 4;
  int arow = wave * 16 + lm;

  union Frag { bf16x16 v; uint4 q[2]; };
  // A fragments (16x32 bf16 layout): V0-3 k=8*lh.. , V4-7 k=16+8*lh..
  Frag fa;
  fa.q[0] = *(const uint4*)&As[arow * DIMS + 8 * lh];
  fa.q[1] = *(const uint4*)&As[arow * DIMS + 16 + 8 * lh];
  bf16x16 a0 = fa.v;
  fa.q[0] = *(const uint4*)&As[arow * DIMS + 32 + 8 * lh];
  fa.q[1] = *(const uint4*)&As[arow * DIMS + 48 + 8 * lh];
  bf16x16 a1 = fa.v;

  const float s2e = 2.0f * LOG2E / EPSR;

  for (int nt = 0; nt < 8; ++nt) {
    int brow = nt * 16 + lm; // column of the output tile = row of P_j
    // B fragments (32x16 bf16 layout): lane N=lm, V0-3 k=8*lh.. , V4-7 k=16+8*lh..
    Frag fb;
    fb.q[0] = *(const uint4*)&Bs[brow * DIMS + 8 * lh];
    fb.q[1] = *(const uint4*)&Bs[brow * DIMS + 16 + 8 * lh];
    bf16x16 b0 = fb.v;
    fb.q[0] = *(const uint4*)&Bs[brow * DIMS + 32 + 8 * lh];
    fb.q[1] = *(const uint4*)&Bs[brow * DIMS + 48 + 8 * lh];
    bf16x16 b1 = fb.v;

    f32x8 acc = {};
    acc = __builtin_amdgcn_wmma_f32_16x16x32_bf16(false, a0, false, b0,
                                                  (short)0, acc, false, false);
    acc = __builtin_amdgcn_wmma_f32_16x16x32_bf16(false, a1, false, b1,
                                                  (short)0, acc, false, false);
    float cj = cB[brow];
    #pragma unroll
    for (int r = 0; r < 8; ++r) {
      int rl = r + 8 * lh; // local row within 16x16 tile
      float x = fmaf(acc[r], s2e, cA[wave * 16 + rl] + cj + SHIFT2);
      // bounded exponent (~[-10,+6]) -> raw v_exp_f32, no denorm fixup needed
      stage[wave][rl * 16 + lm] = f2bf(__builtin_amdgcn_exp2f(x));
    }
    // coalesced bf16 stores: lane -> (row = lane>>1, 8-half segment = lane&1)
    int srow = lane >> 1, sseg = lane & 1;
    uint4 pk = *(const uint4*)&stage[wave][srow * 16 + 8 * sseg];
    size_t g = (size_t)(i0 + wave * 16 + srow) * NPART + (size_t)(j0 + nt * 16 + 8 * sseg);
    *(uint4*)&K[g] = pk;
  }
}

// ---------------------------------------------------------------------------
// matvec: y = K * (x / *Sx).  K symmetric -> also covers K^T. Wave per row,
// x async-staged in LDS (32KB), bf16 K streamed with b128 loads (L2-resident).
// ---------------------------------------------------------------------------
__global__ void __launch_bounds__(256) dpf_matvec(const unsigned short* __restrict__ K,
                                                  const float* __restrict__ x,
                                                  const float* __restrict__ Sx,
                                                  float* __restrict__ y) {
  __shared__ __align__(16) float xs[NPART];
  int t = threadIdx.x;
  #pragma unroll
  for (int it = 0; it < 8; ++it) {
    int idx = (it * 256 + t) * 4;
    async_copy_b128(&xs[idx], x, (unsigned)idx * 4u);
  }
  async_wait0();
  __syncthreads();
  int wave = t >> 5, lane = t & 31;
  int row = blockIdx.x * 8 + wave;
  const unsigned short* Kr = K + (size_t)row * NPART;
  float inv = 1.0f / *Sx;
  float acc = 0.0f;
  for (int it = 0; it < 32; ++it) {
    int base = it * 256 + lane * 8;
    __builtin_prefetch((const void*)(Kr + base + 2048), 0, 1);
    uint4 kq = *(const uint4*)&Kr[base];
    float4 xa = *(const float4*)&xs[base];
    float4 xb = *(const float4*)&xs[base + 4];
    acc = fmaf(bflo(kq.x), xa.x, acc); acc = fmaf(bfhi(kq.x), xa.y, acc);
    acc = fmaf(bflo(kq.y), xa.z, acc); acc = fmaf(bfhi(kq.y), xa.w, acc);
    acc = fmaf(bflo(kq.z), xb.x, acc); acc = fmaf(bfhi(kq.z), xb.y, acc);
    acc = fmaf(bflo(kq.w), xb.z, acc); acc = fmaf(bfhi(kq.w), xb.w, acc);
  }
  #pragma unroll
  for (int off = 16; off; off >>= 1) acc += __shfl_down(acc, off, 32);
  if (lane == 0) y[row] = acc * inv;
}

// Deterministic tree-reduce: *S = sum(y)  (no float atomics anywhere)
__global__ void __launch_bounds__(256) dpf_reduce(const float* __restrict__ y,
                                                  float* __restrict__ S) {
  __shared__ float sm[256];
  int t = threadIdx.x;
  float a = 0.0f;
  for (int i = t; i < NPART; i += 256) a += y[i];
  sm[t] = a; __syncthreads();
  for (int s = 128; s > 0; s >>= 1) { if (t < s) sm[t] += sm[t + s]; __syncthreads(); }
  if (t == 0) *S = sm[0];
}

// ---------------------------------------------------------------------------
// sample: row i of row-stochastic T is prop. to K_ij * v_j  (u_i cancels).
// One sweep: row sum + 32 chunk sums -> inverse-CDF chunk search -> shfl
// prefix scan within chunk -> gather particles[idx] into out row.
// ---------------------------------------------------------------------------
__global__ void __launch_bounds__(256) dpf_sample(const unsigned short* __restrict__ K,
                                                  const float* __restrict__ vvec,
                                                  const float* __restrict__ P,
                                                  float* __restrict__ out) {
  __shared__ __align__(16) float vs[NPART];
  __shared__ float chunks[8][32];
  int t = threadIdx.x;
  #pragma unroll
  for (int it = 0; it < 8; ++it) {
    int idx = (it * 256 + t) * 4;
    async_copy_b128(&vs[idx], vvec, (unsigned)idx * 4u);
  }
  async_wait0();
  __syncthreads();
  int wave = t >> 5, lane = t & 31;
  int row = blockIdx.x * 8 + wave;
  const unsigned short* Kr = K + (size_t)row * NPART;

  float rowSum = 0.0f;
  for (int it = 0; it < 32; ++it) {
    int base = it * 256 + lane * 8;
    uint4 kq = *(const uint4*)&Kr[base];
    float4 xa = *(const float4*)&vs[base];
    float4 xb = *(const float4*)&vs[base + 4];
    float part = bflo(kq.x) * xa.x + bfhi(kq.x) * xa.y +
                 bflo(kq.y) * xa.z + bfhi(kq.y) * xa.w +
                 bflo(kq.z) * xb.x + bfhi(kq.z) * xb.y +
                 bflo(kq.w) * xb.z + bfhi(kq.w) * xb.w;
    #pragma unroll
    for (int off = 16; off; off >>= 1) part += __shfl_xor(part, off, 32);
    if (lane == 0) chunks[wave][it] = part;
    rowSum += part;
  }
  // per-row uniform (hash; RNG stream can't bit-match JAX threefry anyway)
  unsigned h = (unsigned)row * 2654435761u ^ 42u;
  h ^= h >> 16; h *= 0x85ebca6bu; h ^= h >> 13; h *= 0xc2b2ae35u; h ^= h >> 16;
  float target = (((h >> 8) + 0.5f) * (1.0f / 16777216.0f)) * rowSum;

  float run = 0.0f, cbase = 0.0f; int cidx = 31; bool found = false;
  for (int it = 0; it < 32; ++it) {
    float cs = chunks[wave][it];
    if (!found && run + cs > target) { cidx = it; cbase = run; found = true; }
    run += cs;
  }
  int base = cidx * 256 + lane * 8;
  uint4 kq = *(const uint4*)&Kr[base];
  float4 xa = *(const float4*)&vs[base];
  float4 xb = *(const float4*)&vs[base + 4];
  float pv[8] = { bflo(kq.x) * xa.x, bfhi(kq.x) * xa.y, bflo(kq.y) * xa.z,
                  bfhi(kq.y) * xa.w, bflo(kq.z) * xb.x, bfhi(kq.z) * xb.y,
                  bflo(kq.w) * xb.z, bfhi(kq.w) * xb.w };
  float ls = 0.0f;
  #pragma unroll
  for (int j = 0; j < 8; ++j) ls += pv[j];
  float inc = ls;
  #pragma unroll
  for (int off = 1; off < 32; off <<= 1) {
    float nv = __shfl_up(inc, off, 32);
    if (lane >= off) inc += nv;
  }
  float excl = inc - ls;
  bool hit = (cbase + excl <= target) && (cbase + inc > target);
  unsigned long long bal = __ballot(hit);
  int src = bal ? (__ffsll((long long)bal) - 1) : 31;
  int jsel = base + 7;
  if (lane == src) {
    float r2 = cbase + excl;
    #pragma unroll
    for (int j = 0; j < 8; ++j) {
      if (r2 + pv[j] > target) { jsel = base + j; break; }
      r2 += pv[j];
    }
  }
  jsel = __shfl(jsel, src, 32);
  float2 val = *(const float2*)&P[(size_t)jsel * DIMS + lane * 2];
  *(float2*)&out[(size_t)row * DIMS + lane * 2] = val;
}

// ---------------------------------------------------------------------------
extern "C" void kernel_launch(void* const* d_in, const int* in_sizes, int n_in,
                              void* d_out, int out_size, void* d_ws, size_t ws_size,
                              hipStream_t stream) {
  const float* particles = (const float*)d_in[0]; // [8192,64] f32
  const float* weights   = (const float*)d_in[1]; // [8192]    f32
  float* out = (float*)d_out;                     // [8192,64] f32

  char* ws = (char*)d_ws;
  unsigned short* K = (unsigned short*)ws;                 // 8192*8192 bf16 = 128 MiB
  size_t off = (size_t)NPART * NPART * 2;
  unsigned short* Pbf = (unsigned short*)(ws + off); off += (size_t)NPART * DIMS * 2;
  float* c = (float*)(ws + off); off += (size_t)NPART * 4;
  float* u = (float*)(ws + off); off += (size_t)NPART * 4;
  float* v = (float*)(ws + off); off += (size_t)NPART * 4;
  float* S = (float*)(ws + off); off += 128 * 4;           // S[0..100]

  dpf_init<<<NPART / 8, 256, 0, stream>>>(particles, weights, Pbf, c, v, S);
  dpf_build_k<<<dim3(NPART / 128, NPART / 128), 256, 0, stream>>>(Pbf, c, K);

  for (int it = 0; it < 50; ++it) {
    dpf_matvec<<<NPART / 8, 256, 0, stream>>>(K, v, &S[2 * it], u);
    dpf_reduce<<<1, 256, 0, stream>>>(u, &S[2 * it + 1]);
    dpf_matvec<<<NPART / 8, 256, 0, stream>>>(K, u, &S[2 * it + 1], v); // K^T == K
    dpf_reduce<<<1, 256, 0, stream>>>(v, &S[2 * it + 2]);
  }
  dpf_sample<<<NPART / 8, 256, 0, stream>>>(K, v, particles, out);
}
